// Attention_82454782149145
// MI455X (gfx1250) — matrix operand order repaired and checked by reading, not compiled, not run on hardware
//
#include <hip/hip_runtime.h>

typedef __attribute__((ext_vector_type(16))) _Float16 v16h;
typedef __attribute__((ext_vector_type(8)))  float    v8f;

#define Bdim 4
#define Ndim 2048
#define Fdim 128
#define Ddim 128
#define Hdim 4
#define NEGV (-9.0e15f)

// ---------------------------------------------------------------------------
// Kernel 1: Wh[h][b] = x[b] @ W[h]   (2048x128 @ 128x128), f16 WMMA, f32 acc.
// One wave (32 lanes) per 16-row M-tile; computes full 16x128 output strip.
// ---------------------------------------------------------------------------
__global__ void k1_wh(const float* __restrict__ x, const float* __restrict__ W,
                      _Float16* __restrict__ whh) {
  const int wid   = blockIdx.x;          // hb*128 + tile
  const int tile  = wid & 127;
  const int hb    = wid >> 7;            // h*B + b
  const int h     = hb >> 2;
  const int b     = hb & 3;
  const int lane  = threadIdx.x;
  const int m     = lane & 15;           // A-matrix M index (both lane halves)
  const int lbase = (lane & 16) ? 8 : 0; // A-matrix K offset for upper lanes
  const int i0    = tile * 16;

  const float* xrow  = x + ((size_t)(b * Ndim + i0 + m)) * Fdim;
  const float* Whead = W + (size_t)h * Fdim * Ddim;

  v8f acc[8] = {};

  for (int k0 = 0; k0 < Fdim; k0 += 32) {
    // A fragment (16x32 f16): lane holds row m, K = {lbase..lbase+7, lbase+16..lbase+23}
    v16h a;
    const float* pa0 = xrow + k0 + lbase;
    const float* pa1 = xrow + k0 + lbase + 16;
#pragma unroll
    for (int e = 0; e < 8; ++e) a[e]     = (_Float16)pa0[e];
#pragma unroll
    for (int e = 0; e < 8; ++e) a[8 + e] = (_Float16)pa1[e];

    // B fragments (32x16 f16 each): lane holds column N = lane&15, K in same order as A
    const float* wrow = Whead + (size_t)k0 * Ddim + (lane & 15);
#pragma unroll
    for (int f = 0; f < 8; ++f) {
      v16h bf;
#pragma unroll
      for (int n = 0; n < 16; ++n) bf[n] = (_Float16)wrow[(size_t)(lbase + (n & 7) + ((n >> 3) << 4)) * Ddim + f * 16];
      acc[f] = __builtin_amdgcn_wmma_f32_16x16x32_f16(false, a, false, bf,
                                                      (short)0, acc[f], false, false);
    }
  }

  // C/D layout: VGPR r, lanes 0-15 -> M=r, lanes 16-31 -> M=8+r; N = lane&15
  const int nout = lane & 15;
  const int mb   = (lane & 16) ? 8 : 0;
#pragma unroll
  for (int f = 0; f < 8; ++f)
#pragma unroll
    for (int r = 0; r < 8; ++r)
      whh[((size_t)hb * Ndim + i0 + mb + r) * Ddim + f * 16 + nout] = (_Float16)acc[f][r];
}

// ---------------------------------------------------------------------------
// Kernel 1b: s1 = Wh . a1 , s2 = Wh . a2  (one thread per (h,b,n) row)
// ---------------------------------------------------------------------------
__global__ void k1b_scores(const _Float16* __restrict__ whh,
                           const float* __restrict__ a_heads,
                           float* __restrict__ s1, float* __restrict__ s2) {
  const int row = blockIdx.x * blockDim.x + threadIdx.x; // hb*N + n
  if (row >= Hdim * Bdim * Ndim) return;
  const int hb = row >> 11;
  const int h  = hb >> 2;
  const _Float16* w = whh + (size_t)row * Ddim;
  const float* a1 = a_heads + (size_t)h * 2 * Ddim;
  const float* a2 = a1 + Ddim;
  float acc1 = 0.f, acc2 = 0.f;
  for (int d = 0; d < Ddim; ++d) {
    const float v = (float)w[d];
    acc1 += v * a1[d];
    acc2 += v * a2[d];
  }
  s1[row] = acc1;
  s2[row] = acc2;
}

// ---------------------------------------------------------------------------
// Kernel 2a: per-row masked softmax stats (max, 1/sum(exp)). One wave per row.
// ---------------------------------------------------------------------------
__global__ void k2a_stats(const float* __restrict__ s1, const float* __restrict__ s2,
                          const int* __restrict__ adj,
                          float* __restrict__ rmax, float* __restrict__ rinv) {
  const int row  = blockIdx.x;            // hb*N + i
  const int lane = threadIdx.x;
  const int hb   = row >> 11;
  const int b    = hb & 3;
  const int i    = row & (Ndim - 1);
  const float s1v = s1[row];
  const float* s2row  = s2  + (size_t)hb * Ndim;
  const int*   adjrow = adj + ((size_t)b * Ndim + i) * Ndim;

  float mx = NEGV;
  for (int j = lane; j < Ndim; j += 32) {
    float v = s1v * s2row[j];
    v = v > 0.f ? v : 0.01f * v;           // leaky_relu
    v = adjrow[j] > 0 ? v : NEGV;          // mask
    mx = fmaxf(mx, v);
  }
#pragma unroll
  for (int off = 16; off > 0; off >>= 1) mx = fmaxf(mx, __shfl_xor(mx, off, 32));

  float sum = 0.f;
  for (int j = lane; j < Ndim; j += 32) {
    float v = s1v * s2row[j];
    v = v > 0.f ? v : 0.01f * v;
    v = adjrow[j] > 0 ? v : NEGV;
    sum += __expf(v - mx);
  }
#pragma unroll
  for (int off = 16; off > 0; off >>= 1) sum += __shfl_xor(sum, off, 32);

  if (lane == 0) { rmax[row] = mx; rinv[row] = 1.0f / sum; }
}

// ---------------------------------------------------------------------------
// Kernel 2b: h' = elu(softmax(e) @ Wh), fused.
// 8 waves / workgroup; each wave owns one 16-row M-tile; all share the 32x128
// Wh_f16 j-tile (8KB), double-buffered in LDS and filled with gfx1250
// global_load_async_to_lds_b128 (ASYNCcnt) so the copy of tile j+1 overlaps
// the 8 WMMAs of tile j. Probs are generated in registers directly into the
// WMMA A-fragment lane layout.
// ---------------------------------------------------------------------------
__global__ void __launch_bounds__(256) k2b_attn(
    const float* __restrict__ s1, const float* __restrict__ s2,
    const float* __restrict__ rmax, const float* __restrict__ rinv,
    const int* __restrict__ adj, const _Float16* __restrict__ whh,
    float* __restrict__ hprime) {
  const int blk   = blockIdx.x;          // hb*16 + super
  const int super = blk & 15;
  const int hb    = blk >> 4;
  const int b     = hb & 3;
  const int t     = threadIdx.x;         // 0..255
  const int w     = t >> 5;              // wave id 0..7
  const int lane  = t & 31;
  const int m     = lane & 15;
  const int lbase = (lane & 16) ? 8 : 0;
  const int i0    = super * 128 + w * 16;
  const int i     = i0 + m;

  const int row = hb * Ndim + i;
  const float s1v = s1[row];
  const float mx  = rmax[row];
  const float inv = rinv[row];
  const float* s2row  = s2  + (size_t)hb * Ndim;
  const int*   adjrow = adj + ((size_t)b * Ndim + i) * Ndim;
  const _Float16* whbase = whh + (size_t)hb * Ndim * Ddim;

  // double-buffered 32x128 f16 tile of Wh (2 x 8KB)
  __shared__ __align__(16) _Float16 ltile[2][32 * Ddim];

  // Each of the 256 threads stages 32B (two b128, offset applies to BOTH
  // the LDS and global addresses per the ISA async pseudocode).
  auto stage = [&](int buf, int j0) {
    const _Float16* g = whbase + (size_t)j0 * Ddim + t * 16;
    unsigned l = (unsigned)(uintptr_t)(&ltile[buf][t * 16]); // low 32b = LDS offset
    asm volatile("global_load_async_to_lds_b128 %0, %1, off\n\t"
                 "global_load_async_to_lds_b128 %0, %1, off offset:16"
                 :: "v"(l), "v"(g) : "memory");
  };

  stage(0, 0);

  v8f acc[8] = {};

  for (int jt = 0; jt < Ndim / 32; ++jt) {
    const int j0  = jt * 32;
    const int buf = jt & 1;

    // my async copies of tile jt have landed
    asm volatile("s_wait_asynccnt 0x0" ::: "memory");
    // (a) everyone's portion of tile jt visible; (b) everyone done reading buf^1
    __syncthreads();
    if (jt + 1 < Ndim / 32) stage(buf ^ 1, j0 + 32);

    // Build A fragment = normalized attention probs for this 16x32 tile.
    // Element idx c*8+e maps to K = lbase + c*16 + e (ISA 16-bit A layout).
    v16h a;
#pragma unroll
    for (int c = 0; c < 2; ++c) {
      const int jb = j0 + lbase + c * 16;
#pragma unroll
      for (int e = 0; e < 8; ++e) {
        const int j = jb + e;
        float v = s1v * s2row[j];
        v = v > 0.f ? v : 0.01f * v;
        const float p = (adjrow[j] > 0) ? __expf(v - mx) * inv : 0.f;
        a[c * 8 + e] = (_Float16)p;
      }
    }

    // B fragments from LDS: lane holds column N = lane&15, K in same order as A
    const _Float16* wl = &ltile[buf][lane & 15];
#pragma unroll
    for (int f = 0; f < 8; ++f) {
      v16h bf;
#pragma unroll
      for (int s = 0; s < 16; ++s) bf[s] = wl[(lbase + (s & 7) + ((s >> 3) << 4)) * Ddim + f * 16];
      acc[f] = __builtin_amdgcn_wmma_f32_16x16x32_f16(false, a, false, bf,
                                                      (short)0, acc[f], false, false);
    }
  }

  // elu + store (C/D layout: VGPR r -> M = r + 8*(lane>=16), N = lane&15)
  const int nout = lane & 15;
  const int mb   = (lane & 16) ? 8 : 0;
#pragma unroll
  for (int f = 0; f < 8; ++f)
#pragma unroll
    for (int r = 0; r < 8; ++r) {
      float v = acc[f][r];
      v = v > 0.f ? v : (__expf(v) - 1.f);
      hprime[((size_t)hb * Ndim + i0 + mb + r) * Ddim + f * 16 + nout] = v;
    }
}

// ---------------------------------------------------------------------------
// Kernel 3: y = cat @ fc_w + fc_b + x ; layernorm over last dim (128).
// One 128-thread block per (b,n) row; thread d owns output column d.
// ---------------------------------------------------------------------------
__global__ void k3_fc_ln(const float* __restrict__ hprime, const float* __restrict__ fc_w,
                         const float* __restrict__ fc_b, const float* __restrict__ x,
                         float* __restrict__ out) {
  const int bn = blockIdx.x;          // b*N + n
  const int d  = threadIdx.x;         // 0..127
  const int b  = bn >> 11;
  const int n  = bn & (Ndim - 1);

  float acc = 0.f;
#pragma unroll 4
  for (int k = 0; k < Hdim * Ddim; ++k) {
    const int h  = k >> 7;
    const int dp = k & 127;
    const float cv = hprime[(((size_t)(h * Bdim + b)) * Ndim + n) * Ddim + dp];
    acc += cv * fc_w[(size_t)k * Ddim + d];
  }
  const float y = acc + fc_b[d] + x[(size_t)bn * Fdim + d];

  __shared__ float rsum[128];
  __shared__ float rsq[128];
  rsum[d] = y;
  rsq[d]  = y * y;
  __syncthreads();
#pragma unroll
  for (int s = 64; s > 0; s >>= 1) {
    if (d < s) { rsum[d] += rsum[d + s]; rsq[d] += rsq[d + s]; }
    __syncthreads();
  }
  const float mu  = rsum[0] * (1.0f / 128.0f);
  const float var = rsq[0] * (1.0f / 128.0f) - mu * mu;
  out[(size_t)bn * Fdim + d] = (y - mu) * rsqrtf(var + 1e-5f);
}

// ---------------------------------------------------------------------------
extern "C" void kernel_launch(void* const* d_in, const int* in_sizes, int n_in,
                              void* d_out, int out_size, void* d_ws, size_t ws_size,
                              hipStream_t stream) {
  const float* x       = (const float*)d_in[0];
  const int*   adj     = (const int*)  d_in[1];
  const float* W_heads = (const float*)d_in[2];
  const float* a_heads = (const float*)d_in[3];
  const float* fc_w    = (const float*)d_in[4];
  const float* fc_b    = (const float*)d_in[5];
  float* out = (float*)d_out;

  char* ws = (char*)d_ws;
  size_t off = 0;
  _Float16* whh = (_Float16*)(ws + off); off += (size_t)Hdim * Bdim * Ndim * Ddim * 2;
  float* s1     = (float*)(ws + off);    off += (size_t)Hdim * Bdim * Ndim * 4;
  float* s2     = (float*)(ws + off);    off += (size_t)Hdim * Bdim * Ndim * 4;
  float* rmax   = (float*)(ws + off);    off += (size_t)Hdim * Bdim * Ndim * 4;
  float* rinv   = (float*)(ws + off);    off += (size_t)Hdim * Bdim * Ndim * 4;
  float* hprime = (float*)(ws + off);    off += (size_t)Hdim * Bdim * Ndim * Ddim * 4;
  (void)ws_size; (void)in_sizes; (void)n_in; (void)out_size;

  k1_wh     <<<Hdim * Bdim * (Ndim / 16), 32, 0, stream>>>(x, W_heads, whh);
  k1b_scores<<<(Hdim * Bdim * Ndim) / 256, 256, 0, stream>>>(whh, a_heads, s1, s2);
  k2a_stats <<<Hdim * Bdim * Ndim, 32, 0, stream>>>(s1, s2, adj, rmax, rinv);
  k2b_attn  <<<Hdim * Bdim * (Ndim / 128), 256, 0, stream>>>(s1, s2, rmax, rinv, adj, whh, hprime);
  k3_fc_ln  <<<Bdim * Ndim, 128, 0, stream>>>(hprime, fc_w, fc_b, x, out);
}
